// NeRFRenderer_84112639525772
// MI455X (gfx1250) — compile-verified, matrix operand order checked by reading, not run on hardware
//
#include <hip/hip_runtime.h>

typedef __attribute__((ext_vector_type(16))) _Float16 v16h;
typedef __attribute__((ext_vector_type(8)))  float    v8f;

#define WMMA_F16(A, B, C) \
    __builtin_amdgcn_wmma_f32_16x16x32_f16(false, (A), false, (B), (short)0, (C), false, false)

#define L2E 1.4426950408889634f   // log2(e)
#define LN2 0.6931471805599453f   // ln(2)

__device__ __forceinline__ float exp2_fast(float x) { return __builtin_amdgcn_exp2f(x); }
__device__ __forceinline__ float log2_fast(float x) { return __builtin_amdgcn_logf(x); }
__device__ __forceinline__ float rcp_fast(float x)  { return __builtin_amdgcn_rcpf(x); }

__device__ __forceinline__ float softplus_f(float x) {
    // ln(1+e^x) = ln2 * log2(1 + 2^(x*log2e));  for large x just x
    return x > 20.f ? x : LN2 * log2_fast(1.f + exp2_fast(x * L2E));
}
__device__ __forceinline__ float sigmoid_f(float x) {
    return rcp_fast(1.f + exp2_fast(-x * L2E));
}

__global__ __launch_bounds__(256)
void nerf_wmma_kernel(const float* __restrict__ rays_o,
                      const float* __restrict__ rays_d,
                      const float* __restrict__ W1,   const float* __restrict__ b1,
                      const float* __restrict__ Wsig, const float* __restrict__ Wsig_d,
                      const float* __restrict__ Wc1,  const float* __restrict__ bc1,
                      const float* __restrict__ Wc2,  const float* __restrict__ Wc2_d,
                      const int*   __restrict__ num_steps,
                      float* __restrict__ out, int N)
{
    const int tid  = blockIdx.x * blockDim.x + threadIdx.x;
    const int lane = threadIdx.x & 31;
    const int wave = tid >> 5;
    const int m    = lane & 15;        // row / ray-slot within tile
    const bool hi  = (lane & 16) != 0; // which K-half this lane holds
    const int rayBase = wave * 16;

    int ray = rayBase + m;
    if (ray >= N) ray = N - 1;

    // ---------------- constant A-matrix fragments (transposed weights) -------
    // A element e at this lane is (M = m, K = kidx(e)),
    // kidx(e) = (e<8 ? e : e+8) + (hi ? 8 : 0)   [16-bit 16x32 WMMA layout]
    v16h A1t0, A1t1, Ac1t0, Ac1t1, Asig, Acol;
#pragma unroll
    for (int e = 0; e < 16; ++e) {
        const int K = (e < 8 ? e : e + 8) + (hi ? 8 : 0);
        // trunk layer-1: A = W1^T (hidden x 3), K padded to 32
        float a0 = (K < 3) ? W1[K * 32 + m]        : 0.f;
        float a1 = (K < 3) ? W1[K * 32 + (m + 16)] : 0.f;
        A1t0[e] = (_Float16)a0;
        A1t1[e] = (_Float16)a1;
        // color layer-1: A = Wc1^T (hidden x 6)
        float c0 = (K < 6) ? Wc1[K * 32 + m]        : 0.f;
        float c1 = (K < 6) ? Wc1[K * 32 + (m + 16)] : 0.f;
        Ac1t0[e] = (_Float16)c0;
        Ac1t1[e] = (_Float16)c1;
        // sigma heads: row 0 = Wsig^T, row 1 = Wsig_d^T (K = hidden, all valid)
        float sv = (m == 0) ? Wsig[K] : ((m == 1) ? Wsig_d[K] : 0.f);
        Asig[e] = (_Float16)sv;
        // color heads: rows 0..2 = Wc2^T, rows 3..5 = Wc2_d^T  (Wc2 is [32,3])
        float cv = (m < 3) ? Wc2[K * 3 + m] : ((m < 6) ? Wc2_d[K * 3 + (m - 3)] : 0.f);
        Acol[e] = (_Float16)cv;
    }
    // Biases broadcast into WMMA C operands: C element i -> row M = i + 8*hi,
    // same value for every column (lane). Loop-invariant v8f per tile.
    v8f CbT0, CbT1, CbC0, CbC1;
#pragma unroll
    for (int i = 0; i < 8; ++i) {
        const int h0 = i + (hi ? 8 : 0);
        CbT0[i] = b1[h0];      CbT1[i] = b1[h0 + 16];
        CbC0[i] = bc1[h0];     CbC1[i] = bc1[h0 + 16];
    }

    // ---------------- per-ray setup: normalize dir, AABB slab near/far -------
    float ox = rays_o[ray * 3 + 0], oy = rays_o[ray * 3 + 1], oz = rays_o[ray * 3 + 2];
    float dx = rays_d[ray * 3 + 0], dy = rays_d[ray * 3 + 1], dz = rays_d[ray * 3 + 2];
    {
        float il = 1.f / sqrtf(dx * dx + dy * dy + dz * dz);
        dx *= il; dy *= il; dz *= il;
    }
    float near, far;
    {
        float ivx = 1.f / dx, ivy = 1.f / dy, ivz = 1.f / dz;
        float t1x = (-1.f - ox) * ivx, t2x = (1.f - ox) * ivx;
        float t1y = (-1.f - oy) * ivy, t2y = (1.f - oy) * ivy;
        float t1z = (-1.f - oz) * ivz, t2z = (1.f - oz) * ivz;
        near = fmaxf(fmaxf(fminf(t1x, t2x), fminf(t1y, t2y)), fminf(t1z, t2z));
        far  = fminf(fminf(fmaxf(t1x, t2x), fmaxf(t1y, t2y)), fmaxf(t1z, t2z));
        near = fmaxf(near, 0.2f);
        far  = fmaxf(far, near + 1e-6f);
    }

    const int   T      = *num_steps;
    const float invTm1 = 1.f / (float)(T - 1);
    const float fn     = far - near;
    const float dstep  = fn * invTm1;       // z_vals[t+1]-z_vals[t]
    const float dlast  = fn / (float)T;     // sample_dist for last delta

    // B operands persist across the loop; zero elements are written exactly
    // once. Direction half of the color features is loop-invariant.
    // Lanes with hi carry K>=8 of the padded inputs -> must stay zero.
    v16h Bm = {};
    v16h Bc = {};
    Bc[3] = (_Float16)(hi ? 0.f : dx);
    Bc[4] = (_Float16)(hi ? 0.f : dy);
    Bc[5] = (_Float16)(hi ? 0.f : dz);

    // composite accumulators (main + dummy); lanes>=16 compute junk, discarded
    float Tc = 1.f,  ws = 0.f,  dep = 0.f,  im0 = 0.f, im1 = 0.f, im2 = 0.f;
    float Td = 1.f,  wsd = 0.f, depd = 0.f, id0 = 0.f, id1 = 0.f, id2 = 0.f;

    for (int t = 0; t < T; ++t) {
        const float z01 = (float)t * invTm1;
        const float zv  = near + fn * z01;
        const float px = fminf(fmaxf(ox + dx * zv, -1.f), 1.f);
        const float py = fminf(fmaxf(oy + dy * zv, -1.f), 1.f);
        const float pz = fminf(fmaxf(oz + dz * zv, -1.f), 1.f);
        // per-lane select (no EXEC divergence): hi lanes must contribute zero
        const _Float16 qx = (_Float16)(hi ? 0.f : px);
        const _Float16 qy = (_Float16)(hi ? 0.f : py);
        const _Float16 qz = (_Float16)(hi ? 0.f : pz);
        Bm[0] = qx; Bm[1] = qy; Bm[2] = qz;
        Bc[0] = qx; Bc[1] = qy; Bc[2] = qz;

        // trunk: h^T = relu(W1^T X^T + b1), bias folded into WMMA C operand
        v8f D0 = WMMA_F16(A1t0, Bm, CbT0);
        v8f D1 = WMMA_F16(A1t1, Bm, CbT1);
        v16h Bh;
#pragma unroll
        for (int i = 0; i < 8; ++i) {
            Bh[i]     = (_Float16)fmaxf(D0[i], 0.f);
            Bh[i + 8] = (_Float16)fmaxf(D1[i], 0.f);
        }
        v8f Z0 = {};
        v8f Ds = WMMA_F16(Asig, Bh, Z0);        // rows: 0=sigma, 1=sigma_dummy

        // color trunk: hc^T = relu(Wc1^T feat^T + bc1)
        v8f C0 = WMMA_F16(Ac1t0, Bc, CbC0);
        v8f C1 = WMMA_F16(Ac1t1, Bc, CbC1);
        v16h Bhc;
#pragma unroll
        for (int i = 0; i < 8; ++i) {
            Bhc[i]     = (_Float16)fmaxf(C0[i], 0.f);
            Bhc[i + 8] = (_Float16)fmaxf(C1[i], 0.f);
        }
        v8f Z1 = {};
        v8f Dc = WMMA_F16(Acol, Bhc, Z1);       // rows 0..2 rgb, 3..5 rgb_dummy

        // ---- composite (per lane; lane r<16 owns ray r) ----
        const float delta = (t == T - 1) ? dlast : dstep;
        const float dl2   = -delta * L2E;       // fold exp->exp2 conversion
        const float sg  = softplus_f(Ds[0]);
        const float sgd = softplus_f(Ds[1]);
        const float a   = 1.f - exp2_fast(dl2 * sg);   // 1 - e^(-delta*sigma)
        const float ad  = 1.f - exp2_fast(dl2 * sgd);
        const float w   = a  * Tc;
        const float wd  = ad * Td;
        Tc *= (1.f - a  + 1e-15f);
        Td *= (1.f - ad + 1e-15f);
        ws  += w;   dep  += w  * z01;
        wsd += wd;  depd += wd * z01;
        im0 += w  * sigmoid_f(Dc[0]);
        im1 += w  * sigmoid_f(Dc[1]);
        im2 += w  * sigmoid_f(Dc[2]);
        id0 += wd * sigmoid_f(Dc[3]);
        id1 += wd * sigmoid_f(Dc[4]);
        id2 += wd * sigmoid_f(Dc[5]);
    }

    // pack: image(3) | depth | ws | image_d(3) | depth_d   (bg_color = 1)
    if (!hi && (rayBase + m) < N) {
        float* o9 = out + (size_t)(rayBase + m) * 9;
        const float bg  = 1.f - ws;
        const float bgd = 1.f - wsd;
        o9[0] = im0 + bg;  o9[1] = im1 + bg;  o9[2] = im2 + bg;
        o9[3] = dep;
        o9[4] = ws;
        o9[5] = id0 + bgd; o9[6] = id1 + bgd; o9[7] = id2 + bgd;
        o9[8] = depd;
    }
}

extern "C" void kernel_launch(void* const* d_in, const int* in_sizes, int n_in,
                              void* d_out, int out_size, void* d_ws, size_t ws_size,
                              hipStream_t stream) {
    const float* rays_o = (const float*)d_in[0];
    const float* rays_d = (const float*)d_in[1];
    const float* W1     = (const float*)d_in[2];
    const float* b1     = (const float*)d_in[3];
    const float* Wsig   = (const float*)d_in[4];
    const float* Wsig_d = (const float*)d_in[5];
    const float* Wc1    = (const float*)d_in[6];
    const float* bc1    = (const float*)d_in[7];
    const float* Wc2    = (const float*)d_in[8];
    const float* Wc2_d  = (const float*)d_in[9];
    const int*   nsteps = (const int*)d_in[10];

    const int N      = in_sizes[0] / 3;             // 32768 rays
    const int waves  = (N + 15) / 16;               // 16 rays per wave
    const int thr    = waves * 32;
    const int block  = 256;                         // 8 waves / block
    const int grid   = (thr + block - 1) / block;

    nerf_wmma_kernel<<<grid, block, 0, stream>>>(
        rays_o, rays_d, W1, b1, Wsig, Wsig_d, Wc1, bc1, Wc2, Wc2_d,
        nsteps, (float*)d_out, N);
}
